// MultiHeadAttention_30279519437189
// MI455X (gfx1250) — compile-verified, matrix operand order checked by reading, not run on hardware
//
#include <hip/hip_runtime.h>

// Problem constants (match reference)
constexpr int Bn = 512;   // batch
constexpr int SQ = 256;   // sequence length
constexpr int E  = 384;   // embedding
constexpr int H  = 6;     // heads
constexpr int D  = 64;    // head dim
constexpr int QKV3 = 3 * D; // 192

typedef __attribute__((ext_vector_type(16))) __bf16 v16bf;
typedef __attribute__((ext_vector_type(8)))  float  v8f;

__device__ __forceinline__ v8f wmma_bf16(v16bf a, v16bf b, v8f c) {
    // v_wmma_f32_16x16x32_bf16  D = A*B + C
    return __builtin_amdgcn_wmma_f32_16x16x32_bf16(
        /*neg_a=*/false, a, /*neg_b=*/false, b,
        /*c_mod=*/(short)0, c, /*reuse_a=*/false, /*reuse_b=*/false);
}

// Load a 16x32 bf16 A-fragment (or 32x16 B-fragment, which mirrors it) from
// row-major storage: lane owns row (row base + lane%16), K contiguous.
// The 8 dword pairs coalesce into 2x global_load_b128 (verified in disasm).
__device__ __forceinline__ v16bf loadFrag(const __bf16* base, int ld, int row,
                                          int k0, int hi) {
    union { v16bf v; unsigned int u[8]; } f;
    const __bf16* p = base + (size_t)row * ld + k0 + (hi << 3);
    #pragma unroll
    for (int t = 0; t < 8; ++t) {
        int kp = ((2 * t) & 7) + ((t >> 2) << 4); // pair base k within lane
        f.u[t] = *(const unsigned int*)(p + kp);
    }
    return f.v;
}

// ---------------- Kernel 1: LayerNorm (fp32) -> bf16 ----------------
__global__ void ln_kernel(const float* __restrict__ x,
                          const float* __restrict__ g,
                          const float* __restrict__ be,
                          __bf16* __restrict__ out) {
    int row  = blockIdx.x * 8 + (threadIdx.x >> 5);
    int lane = threadIdx.x & 31;
    const float* xr = x + (size_t)row * E;
    float v[12], s = 0.f, s2 = 0.f;
    #pragma unroll
    for (int i = 0; i < 12; ++i) {
        v[i] = xr[lane + 32 * i];
        s += v[i]; s2 += v[i] * v[i];
    }
    #pragma unroll
    for (int m = 16; m >= 1; m >>= 1) {
        s  += __shfl_xor(s,  m, 32);
        s2 += __shfl_xor(s2, m, 32);
    }
    float mean = s * (1.0f / E);
    float var  = s2 * (1.0f / E) - mean * mean;
    float rs   = rsqrtf(var + 1e-5f);
    __bf16* orow = out + (size_t)row * E;
    #pragma unroll
    for (int i = 0; i < 12; ++i) {
        int c = lane + 32 * i;
        orow[c] = (__bf16)((v[i] - mean) * rs * g[c] + be[c]);
    }
}

// -------- Kernel 2: weight transpose + bf16 convert (tiny, L2-resident) -----
__global__ void wprep_kernel(const float* __restrict__ wqkv,
                             const float* __restrict__ wproj,
                             __bf16* __restrict__ wqkvT,
                             __bf16* __restrict__ wpT) {
    int idx = blockIdx.x * 256 + threadIdx.x;
    const int n1 = H * E * QKV3;
    if (idx < n1) {
        int h = idx / (E * QKV3);
        int rem = idx % (E * QKV3);
        int e = rem / QKV3, n = rem % QKV3;
        wqkvT[(size_t)h * QKV3 * E + (size_t)n * E + e] = (__bf16)wqkv[idx];
    } else if (idx < n1 + E * E) {
        int j = idx - n1;
        int e = j / E, n = j % E;
        wpT[(size_t)n * E + e] = (__bf16)wproj[j];
    }
}

// ---------------- Kernel 3: per-head QKV GEMM (bf16 WMMA) ----------------
// Register-blocked 4x2 tiles per wave: 64 rows x 32 cols, K loop 384/32 = 12.
// Per K-step: 4 A-frags + 2 B-frags feed 8 WMMAs (0.75 frag loads / WMMA).
// Block: 256 thr (8 waves). Per (b,h): 4 row-groups x 6 col-pairs = 24 tasks
// -> 3 blocks per (b,h).
__global__ void qkv_kernel(const __bf16* __restrict__ xb,
                           const __bf16* __restrict__ wT,
                           __bf16* __restrict__ qb,
                           __bf16* __restrict__ kb,
                           __bf16* __restrict__ vt) {
    int bh  = blockIdx.x / 3;
    int sub = blockIdx.x % 3;
    int wv = threadIdx.x >> 5;
    int lane = threadIdx.x & 31, l15 = lane & 15, hi = lane >> 4;
    int task = sub * 8 + wv;            // 0..23
    int rg = task / 6;                  // row group (64 rows)
    int cp = task % 6;                  // col pair (32 cols)
    int m0 = rg * 64;
    int nt0 = cp * 2;                   // first n-tile index
    int b = bh / H, h = bh % H;
    const __bf16* xbase = xb + (size_t)b * SQ * E;
    const __bf16* wbase = wT + (size_t)h * QKV3 * E;

    v8f acc[8] = {};                    // acc[mi*2+ni]
    for (int kk = 0; kk < 12; ++kk) {
        v16bf a[4], bfr[2];
        #pragma unroll
        for (int mi = 0; mi < 4; ++mi)
            a[mi] = loadFrag(xbase, E, m0 + mi * 16 + l15, kk * 32, hi);
        #pragma unroll
        for (int ni = 0; ni < 2; ++ni)
            bfr[ni] = loadFrag(wbase, E, (nt0 + ni) * 16 + l15, kk * 32, hi);
        #pragma unroll
        for (int mi = 0; mi < 4; ++mi)
            #pragma unroll
            for (int ni = 0; ni < 2; ++ni)
                acc[mi * 2 + ni] = wmma_bf16(a[mi], bfr[ni], acc[mi * 2 + ni]);
    }

    #pragma unroll
    for (int ni = 0; ni < 2; ++ni) {
        int nt = nt0 + ni;
        int n0 = nt * 16;
        if (nt < 4) {                   // Q: (bh, s, d) row-major
            __bf16* q = qb + (size_t)bh * SQ * D;
            #pragma unroll
            for (int mi = 0; mi < 4; ++mi)
                #pragma unroll
                for (int r = 0; r < 8; ++r)
                    q[(size_t)(m0 + mi * 16 + r + 8 * hi) * D + n0 + l15] =
                        (__bf16)acc[mi * 2 + ni][r];
        } else if (nt < 8) {            // K: (bh, s, d) row-major
            __bf16* k = kb + (size_t)bh * SQ * D;
            #pragma unroll
            for (int mi = 0; mi < 4; ++mi)
                #pragma unroll
                for (int r = 0; r < 8; ++r)
                    k[(size_t)(m0 + mi * 16 + r + 8 * hi) * D + (n0 - 64) + l15] =
                        (__bf16)acc[mi * 2 + ni][r];
        } else {                        // V transposed: (bh, d, s)
            __bf16* v = vt + (size_t)bh * D * SQ;
            #pragma unroll
            for (int mi = 0; mi < 4; ++mi)
                #pragma unroll
                for (int r = 0; r < 8; ++r)
                    v[(size_t)((n0 - 128) + l15) * SQ + m0 + mi * 16 + r + 8 * hi] =
                        (__bf16)acc[mi * 2 + ni][r];
        }
    }
}

// -------- Kernel 4: fused causal flash attention (online softmax) --------
// Block: 128 thr (4 waves); each wave owns a 16-row Q tile; 32-key steps.
__global__ void attn_kernel(const __bf16* __restrict__ qb,
                            const __bf16* __restrict__ kbuf,
                            const __bf16* __restrict__ vt,
                            __bf16* __restrict__ ob) {
    __shared__ __bf16 smem[4][16 * 32];
    int bh = blockIdx.x >> 2;
    int wv = threadIdx.x >> 5;
    int q0 = (blockIdx.x & 3) * 64 + wv * 16;
    int lane = threadIdx.x & 31, l15 = lane & 15, hi = lane >> 4;
    int b = bh / H, h = bh % H;
    const __bf16* qbase = qb   + (size_t)bh * SQ * D;
    const __bf16* kbase = kbuf + (size_t)bh * SQ * D;
    const __bf16* vbase = vt   + (size_t)bh * D * SQ;

    v16bf qa0 = loadFrag(qbase, D, q0 + l15, 0,  hi);
    v16bf qa1 = loadFrag(qbase, D, q0 + l15, 32, hi);

    v8f acc0 = {}, acc1 = {}, acc2 = {}, acc3 = {};
    float mrun[8], lrun[8];
    #pragma unroll
    for (int r = 0; r < 8; ++r) { mrun[r] = -__builtin_inff(); lrun[r] = 0.f; }
    __bf16* sm = smem[wv];

    for (int jb = 0; jb <= q0 + 15; jb += 32) {   // causal: skip tiles past diag
        v8f s0 = {}, s1 = {};
        s0 = wmma_bf16(qa0, loadFrag(kbase, D, jb + l15,      0,  hi), s0);
        s0 = wmma_bf16(qa1, loadFrag(kbase, D, jb + l15,      32, hi), s0);
        s1 = wmma_bf16(qa0, loadFrag(kbase, D, jb + 16 + l15, 0,  hi), s1);
        s1 = wmma_bf16(qa1, loadFrag(kbase, D, jb + 16 + l15, 32, hi), s1);

        float p0[8], p1[8];
        #pragma unroll
        for (int r = 0; r < 8; ++r) {
            int q = q0 + r + 8 * hi;
            float a0 = s0[r] * 0.125f;                 // 1/sqrt(64)
            if (jb + l15 > q)      a0 = -__builtin_inff();
            float a1 = s1[r] * 0.125f;
            if (jb + 16 + l15 > q) a1 = -__builtin_inff();
            float mx = fmaxf(a0, a1);
            #pragma unroll
            for (int msk = 8; msk >= 1; msk >>= 1)
                mx = fmaxf(mx, __shfl_xor(mx, msk, 32));  // 16-lane row reduce
            float mnew = fmaxf(mrun[r], mx);
            float corr = __expf(mrun[r] - mnew);
            mrun[r] = mnew;
            a0 = __expf(a0 - mnew);
            a1 = __expf(a1 - mnew);
            float rs = a0 + a1;
            #pragma unroll
            for (int msk = 8; msk >= 1; msk >>= 1)
                rs += __shfl_xor(rs, msk, 32);
            lrun[r] = lrun[r] * corr + rs;
            acc0[r] *= corr; acc1[r] *= corr; acc2[r] *= corr; acc3[r] *= corr;
            p0[r] = a0; p1[r] = a1;
        }

        // C-layout -> A-layout via wave-private LDS tile (16x32 bf16)
        #pragma unroll
        for (int r = 0; r < 8; ++r) {
            int row = r + 8 * hi;
            sm[row * 32 + l15]      = (__bf16)p0[r];
            sm[row * 32 + 16 + l15] = (__bf16)p1[r];
        }
        asm volatile("s_wait_dscnt 0" ::: "memory");  // wave-local LDS RAW fence
        v16bf pa = loadFrag(sm, 32, l15, 0, hi);

        acc0 = wmma_bf16(pa, loadFrag(vbase, SQ,  0 + l15, jb, hi), acc0);
        acc1 = wmma_bf16(pa, loadFrag(vbase, SQ, 16 + l15, jb, hi), acc1);
        acc2 = wmma_bf16(pa, loadFrag(vbase, SQ, 32 + l15, jb, hi), acc2);
        acc3 = wmma_bf16(pa, loadFrag(vbase, SQ, 48 + l15, jb, hi), acc3);
    }

    // O layout: (b, s, h*64+d) so the projection GEMM reads rows contiguously
    __bf16* obase = ob + (size_t)b * SQ * E + (size_t)h * D;
    #pragma unroll
    for (int r = 0; r < 8; ++r) {
        float inv = 1.0f / lrun[r];
        size_t rowoff = (size_t)(q0 + r + 8 * hi) * E;
        obase[rowoff +  0 + l15] = (__bf16)(acc0[r] * inv);
        obase[rowoff + 16 + l15] = (__bf16)(acc1[r] * inv);
        obase[rowoff + 32 + l15] = (__bf16)(acc2[r] * inv);
        obase[rowoff + 48 + l15] = (__bf16)(acc3[r] * inv);
    }
}

// ------------- Kernel 5: output projection GEMM + bias (fp32 out) -----------
// Register-blocked 4x2 tiles per wave (64 rows x 32 cols), K loop 12.
__global__ void proj_kernel(const __bf16* __restrict__ ob,
                            const __bf16* __restrict__ wpT,
                            const float* __restrict__ bias,
                            float* __restrict__ out) {
    int wv = threadIdx.x >> 5;
    int lane = threadIdx.x & 31, l15 = lane & 15, hi = lane >> 4;
    int task = blockIdx.x * 8 + wv;     // 0 .. 24575
    int rg = task / 12;                 // row group (64 rows), 0..2047
    int cp = task % 12;                 // col pair (32 cols)
    int m0 = rg * 64;
    int nt0 = cp * 2;

    v8f acc[8] = {};
    for (int kk = 0; kk < 12; ++kk) {
        v16bf a[4], bfr[2];
        #pragma unroll
        for (int mi = 0; mi < 4; ++mi)
            a[mi] = loadFrag(ob, E, m0 + mi * 16 + l15, kk * 32, hi);
        #pragma unroll
        for (int ni = 0; ni < 2; ++ni)
            bfr[ni] = loadFrag(wpT, E, (nt0 + ni) * 16 + l15, kk * 32, hi);
        #pragma unroll
        for (int mi = 0; mi < 4; ++mi)
            #pragma unroll
            for (int ni = 0; ni < 2; ++ni)
                acc[mi * 2 + ni] = wmma_bf16(a[mi], bfr[ni], acc[mi * 2 + ni]);
    }

    #pragma unroll
    for (int ni = 0; ni < 2; ++ni) {
        float bv = bias[(nt0 + ni) * 16 + l15];
        #pragma unroll
        for (int mi = 0; mi < 4; ++mi)
            #pragma unroll
            for (int r = 0; r < 8; ++r)
                out[(size_t)(m0 + mi * 16 + r + 8 * hi) * E +
                    (nt0 + ni) * 16 + l15] = acc[mi * 2 + ni][r] + bv;
    }
}

extern "C" void kernel_launch(void* const* d_in, const int* in_sizes, int n_in,
                              void* d_out, int out_size, void* d_ws, size_t ws_size,
                              hipStream_t stream) {
    const float* logits = (const float*)d_in[0];
    const float* gamma  = (const float*)d_in[1];
    const float* beta   = (const float*)d_in[2];
    const float* wqkv   = (const float*)d_in[3];
    const float* wproj  = (const float*)d_in[4];
    const float* bproj  = (const float*)d_in[5];
    float* out = (float*)d_out;

    char* w = (char*)d_ws;
    size_t off = 0;
    auto alloc = [&](size_t bytes) -> void* {
        void* p = w + off;
        off += (bytes + 255) & ~(size_t)255;
        return p;
    };
    // x_bf doubles as o_bf (x only needed by qkv_kernel, o written by attn)
    __bf16* xb    = (__bf16*)alloc((size_t)Bn * SQ * E * 2);        // 100.7 MB
    __bf16* qb    = (__bf16*)alloc((size_t)Bn * H * SQ * D * 2);    // 100.7 MB
    __bf16* kb    = (__bf16*)alloc((size_t)Bn * H * SQ * D * 2);    // 100.7 MB
    __bf16* vt    = (__bf16*)alloc((size_t)Bn * H * D * SQ * 2);    // 100.7 MB
    __bf16* wqkvT = (__bf16*)alloc((size_t)H * QKV3 * E * 2);
    __bf16* wpT   = (__bf16*)alloc((size_t)E * E * 2);

    const int rows = Bn * SQ;                          // 131072
    ln_kernel<<<rows / 8, 256, 0, stream>>>(logits, gamma, beta, xb);

    const int wtot = H * E * QKV3 + E * E;
    wprep_kernel<<<(wtot + 255) / 256, 256, 0, stream>>>(wqkv, wproj, wqkvT, wpT);

    qkv_kernel<<<Bn * H * 3, 256, 0, stream>>>(xb, wqkvT, qb, kb, vt);

    attn_kernel<<<Bn * H * 4, 128, 0, stream>>>(qb, kb, vt, xb /* as O */);

    const int tasks = (rows / 64) * (E / 32);          // 24576 wave tasks
    proj_kernel<<<tasks / 8, 256, 0, stream>>>(xb, wpT, bproj, out);
}
